// ImageSNS_52828097741494
// MI455X (gfx1250) — matrix operand order repaired and checked by reading, not compiled, not run on hardware
//
#include <hip/hip_runtime.h>

// ---------------------------------------------------------------------------
// MI455X (gfx1250) CNS RNN:  bf16 WMMA GEMMs + fused recurrent update
//   EXT = X @ W_in^T + b_in           (one big GEMM, hoisted out of scan)
//   28x: P = sigma@(G*E)^T - (sigma@G^T)*state ; update fused in epilogue
//   OUT = state @ W_out^T + b_out
// Data movement: GLOBAL_LOAD_ASYNC_TO_LDS_B128 (ASYNCcnt), 3-stage LDS pipe
// with compile-time stage phases (no runtime stage math in the hot loop).
// ---------------------------------------------------------------------------

typedef __attribute__((ext_vector_type(16))) __bf16 v16bf;
typedef __attribute__((ext_vector_type(8)))  float  v8f;

#define BM 128
#define BN 64
#define BK 32
#define LDA_T (BK + 8)   // padded LDS row stride (elements)
#define LDB_T (BK + 8)
#define NSTAGE 3

#define NHID   2048
#define TSTEPS 28

union FragU { uint4 u[2]; v16bf v; };

// Async direct-to-LDS 16B copy (per lane), GV form with immediate offset:
//   vdst = LDS byte address, vaddr = 64-bit global address. Uses ASYNCcnt.
template<int OFF>
__device__ __forceinline__ void async_cp_b128(unsigned lds_addr, const void* gptr) {
    asm volatile("global_load_async_to_lds_b128 %0, %1, off offset:%2"
                 :: "v"(lds_addr), "v"((unsigned long long)(uintptr_t)gptr), "n"(OFF)
                 : "memory");
}
#define ASYNC_WAIT(n) asm volatile("s_wait_asynccnt " #n ::: "memory")

__device__ __forceinline__ unsigned lds_addr_of(const void* p) {
    return (unsigned)(uintptr_t)p;   // low 32 bits of generic ptr = LDS offset
}

// A fragment (16x32 bf16): lanes 0-15: row M=lane, K 0..7 | 16..23;
// lanes 16-31: same rows, K 8..15 | 24..31.  LDS is [BM][LDA_T] row-major.
__device__ __forceinline__ v16bf ld_a_frag(const __bf16* As, int row_base, int lane) {
    const int row  = row_base + (lane & 15);
    const int koff = (lane < 16) ? 0 : 8;
    const __bf16* p = As + row * LDA_T + koff;
    FragU f;
    f.u[0] = *reinterpret_cast<const uint4*>(p);
    f.u[1] = *reinterpret_cast<const uint4*>(p + 16);
    return f.v;
}

// B fragment (32x16 bf16): lane holds column N=lane&15; lanes 0-15 K 0..7 | 16..23,
// lanes 16-31 K 8..15 | 24..31.  LDS holds B^T as [BN][LDB_T] so K is contiguous.
__device__ __forceinline__ v16bf ld_b_frag(const __bf16* Bs, int col_base, int lane) {
    const int n    = col_base + (lane & 15);
    const int koff = (lane < 16) ? 0 : 8;
    const __bf16* p = Bs + n * LDB_T + koff;
    FragU f;
    f.u[0] = *reinterpret_cast<const uint4*>(p);
    f.u[1] = *reinterpret_cast<const uint4*>(p + 16);
    return f.v;
}

#define WMMA_BF16(d, a, b) \
    d = __builtin_amdgcn_wmma_f32_16x16x32_bf16(false, a, false, b, (short)0, d, false, false)

// ---------------------------------------------------------------------------
// Generic GEMM:  C[M,N] = A[M,K](bf16) * Bt[N,K]^T(bf16) + bias[N]
// 8 wave32 / block, 128x64 tile, BK=32, 3-stage async pipeline, S=3/stage.
// ---------------------------------------------------------------------------
template<int K>
__global__ __launch_bounds__(256) void gemm_bias_kernel(
    const __bf16* __restrict__ A,
    const __bf16* __restrict__ Bt,
    const float*  __restrict__ bias,
    float* __restrict__ C,
    int M, int N)
{
    constexpr int NK = K / BK;
    static_assert(NK >= 4 && "pipeline needs >= 4 K-steps");

    __shared__ __bf16 As[NSTAGE][BM * LDA_T];
    __shared__ __bf16 Bs[NSTAGE][BN * LDB_T];

    const int tid  = threadIdx.x;
    const int lane = tid & 31;
    const int wave = tid >> 5;
    const int wm   = wave & 3;   // 4 M-groups of 32 rows
    const int wn   = wave >> 2;  // 2 N-groups of 32 cols

    const long m0 = (long)blockIdx.y * BM;
    const long n0 = (long)blockIdx.x * BN;

    // cooperative loaders: A = 2x b128 / thread, B = 1x b128 / thread
    const int arow = tid >> 1;
    const int acol = (tid & 1) * 16;
    const int brow = tid >> 2;
    const int bcol = (tid & 3) * 8;
    const __bf16* agp = A  + (m0 + arow) * (long)K + acol;  // group base
    const __bf16* bgp = Bt + (n0 + brow) * (long)K + bcol;  // group base

    unsigned ldsA0[NSTAGE], ldsA1[NSTAGE], ldsB0[NSTAGE];
    #pragma unroll
    for (int s = 0; s < NSTAGE; ++s) {
        ldsA0[s] = lds_addr_of(&As[s][arow * LDA_T + acol]);
        ldsA1[s] = lds_addr_of(&As[s][arow * LDA_T + acol + 8]);
        ldsB0[s] = lds_addr_of(&Bs[s][brow * LDB_T + bcol]);
    }

    v8f acc[2][2] = {};

    // prologue: stage 0 at +0B, stage 1 at +64B (BK*2)
    async_cp_b128<0>        (ldsA0[0], agp);
    async_cp_b128<16>       (ldsA1[0], agp);
    async_cp_b128<0>        (ldsB0[0], bgp);
    async_cp_b128<BK*2>     (ldsA0[1], agp);
    async_cp_b128<BK*2+16>  (ldsA1[1], agp);
    async_cp_b128<BK*2>     (ldsB0[1], bgp);
    ASYNC_WAIT(3);          // stage 0 resident (in-order completion)
    __syncthreads();

// phase-S pipeline body: prefetch stage (S+2)%3 at +((S+2)*64)B from group
// base, compute on stage S, keep 1 stage in flight.
#define GBODY(S, KT)                                                          \
    {                                                                         \
        const int  kt_  = (KT);                                               \
        const bool pre_ = (kt_ + 2) < NK;                                     \
        if (pre_) {                                                           \
            async_cp_b128<((S)+2)*BK*2>     (ldsA0[((S)+2)%3], agp);          \
            async_cp_b128<((S)+2)*BK*2+16>  (ldsA1[((S)+2)%3], agp);          \
            async_cp_b128<((S)+2)*BK*2>     (ldsB0[((S)+2)%3], bgp);          \
        }                                                                     \
        v16bf af0 = ld_a_frag(As[(S)], wm * 32,      lane);                   \
        v16bf af1 = ld_a_frag(As[(S)], wm * 32 + 16, lane);                   \
        v16bf bf0 = ld_b_frag(Bs[(S)], wn * 32,      lane);                   \
        v16bf bf1 = ld_b_frag(Bs[(S)], wn * 32 + 16, lane);                   \
        WMMA_BF16(acc[0][0], af0, bf0);                                       \
        WMMA_BF16(acc[0][1], af0, bf1);                                       \
        WMMA_BF16(acc[1][0], af1, bf0);                                       \
        WMMA_BF16(acc[1][1], af1, bf1);                                       \
        if (pre_) { ASYNC_WAIT(3); } else { ASYNC_WAIT(0); }                  \
        __syncthreads();                                                      \
    }

    int kt = 0;
    for (int g = 0; g < NK / 3; ++g) {
        GBODY(0, kt)
        GBODY(1, kt + 1)
        GBODY(2, kt + 2)
        kt += 3;
        agp += 3 * BK;
        bgp += 3 * BK;
    }
    if (NK % 3 >= 1) GBODY(0, kt)
    if (NK % 3 >= 2) GBODY(1, kt + 1)
#undef GBODY

    // epilogue: C/D layout: lane = N (lane&15), M = vgpr + 8*(lane>=16)
    const int hi = lane >> 4;
    const int ln = lane & 15;
    #pragma unroll
    for (int i = 0; i < 2; ++i) {
        #pragma unroll
        for (int j = 0; j < 2; ++j) {
            const long col = n0 + wn * 32 + j * 16 + ln;
            const long rb  = m0 + wm * 32 + i * 16 + hi * 8;
            const float bv = bias[col];
            #pragma unroll
            for (int v = 0; v < 8; ++v)
                C[(rb + v) * (long)N + col] = acc[i][j][v] + bv;
        }
    }
}

// ---------------------------------------------------------------------------
// Recurrent step: dual-B GEMM (GE^T and G^T share the sigma A tile) with the
// leaky-integrator update fused into the epilogue.  S = 4 asyncs per stage.
// ---------------------------------------------------------------------------
__global__ __launch_bounds__(256) void cns_step_kernel(
    const __bf16* __restrict__ sig_in,    // [B, NH] clip(state)
    const float*  __restrict__ st_in,     // [B, NH]
    const __bf16* __restrict__ GEbf,      // [NH, NH]  (n,k) = G[n][k]*E[n][k]
    const __bf16* __restrict__ Gbf,       // [NH, NH]
    const float*  __restrict__ ext,       // [B*T, NH], row (b*T + t)
    const float*  __restrict__ tau,
    const float*  __restrict__ leak,
    const float*  __restrict__ rest,
    const float*  __restrict__ bias_h,
    float*  __restrict__ st_out,
    __bf16* __restrict__ sig_out,
    __bf16* __restrict__ st_out_bf,       // written only when last != 0
    int t, int last)
{
    constexpr int NK = NHID / BK;   // 64

    __shared__ __bf16 As [NSTAGE][BM * LDA_T];
    __shared__ __bf16 Bs1[NSTAGE][BN * LDB_T];
    __shared__ __bf16 Bs2[NSTAGE][BN * LDB_T];

    const int tid  = threadIdx.x;
    const int lane = tid & 31;
    const int wave = tid >> 5;
    const int wm   = wave & 3;
    const int wn   = wave >> 2;

    const long m0 = (long)blockIdx.y * BM;
    const long n0 = (long)blockIdx.x * BN;

    const int arow = tid >> 1;
    const int acol = (tid & 1) * 16;
    const int brow = tid >> 2;
    const int bcol = (tid & 3) * 8;
    const __bf16* agp  = sig_in + (m0 + arow) * (long)NHID + acol;
    const __bf16* bgp1 = GEbf   + (n0 + brow) * (long)NHID + bcol;
    const __bf16* bgp2 = Gbf    + (n0 + brow) * (long)NHID + bcol;

    unsigned ldsA0[NSTAGE], ldsA1[NSTAGE], ldsB1[NSTAGE], ldsB2[NSTAGE];
    #pragma unroll
    for (int s = 0; s < NSTAGE; ++s) {
        ldsA0[s] = lds_addr_of(&As [s][arow * LDA_T + acol]);
        ldsA1[s] = lds_addr_of(&As [s][arow * LDA_T + acol + 8]);
        ldsB1[s] = lds_addr_of(&Bs1[s][brow * LDB_T + bcol]);
        ldsB2[s] = lds_addr_of(&Bs2[s][brow * LDB_T + bcol]);
    }

    v8f accP[2][2] = {};   // sigma @ GE^T
    v8f accQ[2][2] = {};   // sigma @ G^T

    async_cp_b128<0>        (ldsA0[0], agp);
    async_cp_b128<16>       (ldsA1[0], agp);
    async_cp_b128<0>        (ldsB1[0], bgp1);
    async_cp_b128<0>        (ldsB2[0], bgp2);
    async_cp_b128<BK*2>     (ldsA0[1], agp);
    async_cp_b128<BK*2+16>  (ldsA1[1], agp);
    async_cp_b128<BK*2>     (ldsB1[1], bgp1);
    async_cp_b128<BK*2>     (ldsB2[1], bgp2);
    ASYNC_WAIT(4);
    __syncthreads();

#define CBODY(S, KT)                                                          \
    {                                                                         \
        const int  kt_  = (KT);                                               \
        const bool pre_ = (kt_ + 2) < NK;                                     \
        if (pre_) {                                                           \
            async_cp_b128<((S)+2)*BK*2>     (ldsA0[((S)+2)%3], agp);          \
            async_cp_b128<((S)+2)*BK*2+16>  (ldsA1[((S)+2)%3], agp);          \
            async_cp_b128<((S)+2)*BK*2>     (ldsB1[((S)+2)%3], bgp1);         \
            async_cp_b128<((S)+2)*BK*2>     (ldsB2[((S)+2)%3], bgp2);         \
        }                                                                     \
        v16bf af0 = ld_a_frag(As [(S)], wm * 32,      lane);                  \
        v16bf af1 = ld_a_frag(As [(S)], wm * 32 + 16, lane);                  \
        v16bf ge0 = ld_b_frag(Bs1[(S)], wn * 32,      lane);                  \
        v16bf ge1 = ld_b_frag(Bs1[(S)], wn * 32 + 16, lane);                  \
        v16bf gg0 = ld_b_frag(Bs2[(S)], wn * 32,      lane);                  \
        v16bf gg1 = ld_b_frag(Bs2[(S)], wn * 32 + 16, lane);                  \
        WMMA_BF16(accP[0][0], af0, ge0);                                      \
        WMMA_BF16(accP[0][1], af0, ge1);                                      \
        WMMA_BF16(accP[1][0], af1, ge0);                                      \
        WMMA_BF16(accP[1][1], af1, ge1);                                      \
        WMMA_BF16(accQ[0][0], af0, gg0);                                      \
        WMMA_BF16(accQ[0][1], af0, gg1);                                      \
        WMMA_BF16(accQ[1][0], af1, gg0);                                      \
        WMMA_BF16(accQ[1][1], af1, gg1);                                      \
        if (pre_) { ASYNC_WAIT(4); } else { ASYNC_WAIT(0); }                  \
        __syncthreads();                                                      \
    }

    int kt = 0;
    for (int g = 0; g < NK / 3; ++g) {
        CBODY(0, kt)
        CBODY(1, kt + 1)
        CBODY(2, kt + 2)
        kt += 3;
        agp  += 3 * BK;
        bgp1 += 3 * BK;
        bgp2 += 3 * BK;
    }
    if (NK % 3 >= 1) CBODY(0, kt)
    if (NK % 3 >= 2) CBODY(1, kt + 1)
#undef CBODY

    // fused leaky-integrator update
    const int hi = lane >> 4;
    const int ln = lane & 15;
    #pragma unroll
    for (int i = 0; i < 2; ++i) {
        #pragma unroll
        for (int j = 0; j < 2; ++j) {
            const long col = n0 + wn * 32 + j * 16 + ln;
            const long rb  = m0 + wm * 32 + i * 16 + hi * 8;
            const float tv = tau[col];
            const float lv = leak[col];
            const float rv = rest[col];
            const float bh = bias_h[col];
            #pragma unroll
            for (int v = 0; v < 8; ++v) {
                const long r = rb + v;
                const float s = st_in[r * (long)NHID + col];
                const float e = ext[(r * (long)TSTEPS + t) * (long)NHID + col];
                const float p = accP[i][j][v] - accQ[i][j][v] * s;
                const float nu = s + tv * (lv * (rv - s) + bh + e + p);
                st_out[r * (long)NHID + col] = nu;
                const float sg = fminf(fmaxf(nu, 0.f), 1.f);
                sig_out[r * (long)NHID + col] = (__bf16)sg;
                if (last) st_out_bf[r * (long)NHID + col] = (__bf16)nu;
            }
        }
    }
}

// ---------------------------------------------------------------------------
// Prep kernels (elementwise, bandwidth-trivial)
// ---------------------------------------------------------------------------
__global__ void cvt_bf16_kernel(const float* __restrict__ src, __bf16* __restrict__ dst, long n) {
    long i = (long)blockIdx.x * blockDim.x + threadIdx.x;
    const long stride = (long)gridDim.x * blockDim.x;
    for (; i < n; i += stride) dst[i] = (__bf16)src[i];
}

__global__ void make_ge_kernel(const float* __restrict__ G, const float* __restrict__ E,
                               __bf16* __restrict__ GEbf, __bf16* __restrict__ Gbf, long n) {
    long i = (long)blockIdx.x * blockDim.x + threadIdx.x;
    const long stride = (long)gridDim.x * blockDim.x;
    for (; i < n; i += stride) {
        const float g = G[i];
        GEbf[i] = (__bf16)(g * E[i]);
        Gbf[i]  = (__bf16)g;
    }
}

__global__ void init_state_kernel(const float* __restrict__ s0, float* __restrict__ st,
                                  __bf16* __restrict__ sg, long n) {
    long i = (long)blockIdx.x * blockDim.x + threadIdx.x;
    const long stride = (long)gridDim.x * blockDim.x;
    for (; i < n; i += stride) {
        const float v = s0[i];
        st[i] = v;
        sg[i] = (__bf16)fminf(fmaxf(v, 0.f), 1.f);
    }
}

// ---------------------------------------------------------------------------
extern "C" void kernel_launch(void* const* d_in, const int* in_sizes, int n_in,
                              void* d_out, int out_size, void* d_ws, size_t ws_size,
                              hipStream_t stream)
{
    constexpr int Bv = 1024, Tv = 28, NIN = 1024, NH = 2048, NOUT = 1024;
    const float* X      = (const float*)d_in[0];
    const float* state0 = (const float*)d_in[1];
    const float* W_in   = (const float*)d_in[2];
    const float* b_in   = (const float*)d_in[3];
    const float* W_out  = (const float*)d_in[4];
    const float* b_out  = (const float*)d_in[5];
    const float* G      = (const float*)d_in[6];
    const float* E      = (const float*)d_in[7];
    const float* tau    = (const float*)d_in[8];
    const float* leak   = (const float*)d_in[9];
    const float* rest   = (const float*)d_in[10];
    const float* bias_h = (const float*)d_in[11];

    char* ws = (char*)d_ws;
    size_t off = 0;
    auto carve = [&](size_t bytes) -> char* {
        char* p = ws + off;
        off += (bytes + 255) & ~(size_t)255;
        return p;
    };

    __bf16* Xbf    = (__bf16*)carve((size_t)Bv * Tv * NIN * 2);
    __bf16* Winbf  = (__bf16*)carve((size_t)NH * NIN * 2);
    __bf16* GEbf   = (__bf16*)carve((size_t)NH * NH * 2);
    __bf16* Gbf    = (__bf16*)carve((size_t)NH * NH * 2);
    __bf16* Woutbf = (__bf16*)carve((size_t)NOUT * NH * 2);
    float*  EXT    = (float*) carve((size_t)Bv * Tv * NH * 4);
    float*  stA    = (float*) carve((size_t)Bv * NH * 4);
    float*  stB    = (float*) carve((size_t)Bv * NH * 4);
    __bf16* sgA    = (__bf16*)carve((size_t)Bv * NH * 2);
    __bf16* sgB    = (__bf16*)carve((size_t)Bv * NH * 2);
    __bf16* stFbf  = (__bf16*)carve((size_t)Bv * NH * 2);

    cvt_bf16_kernel  <<<4096, 256, 0, stream>>>(X,     Xbf,    (long)Bv * Tv * NIN);
    cvt_bf16_kernel  <<<2048, 256, 0, stream>>>(W_in,  Winbf,  (long)NH * NIN);
    cvt_bf16_kernel  <<<2048, 256, 0, stream>>>(W_out, Woutbf, (long)NOUT * NH);
    make_ge_kernel   <<<4096, 256, 0, stream>>>(G, E, GEbf, Gbf, (long)NH * NH);
    init_state_kernel<<<2048, 256, 0, stream>>>(state0, stA, sgA, (long)Bv * NH);

    // EXT[b*T+t, :] = X[b,t,:] @ W_in^T + b_in   (state-independent, hoisted)
    gemm_bias_kernel<1024><<<dim3(NH / BN, (Bv * Tv) / BM), 256, 0, stream>>>(
        Xbf, Winbf, b_in, EXT, Bv * Tv, NH);

    // 28 recurrent steps (ping-pong state / sigma buffers)
    for (int t = 0; t < Tv; ++t) {
        const __bf16* si = (t & 1) ? sgB : sgA;
        __bf16*       so = (t & 1) ? sgA : sgB;
        const float*  pi = (t & 1) ? stB : stA;
        float*        po = (t & 1) ? stA : stB;
        cns_step_kernel<<<dim3(NH / BN, Bv / BM), 256, 0, stream>>>(
            si, pi, GEbf, Gbf, EXT, tau, leak, rest, bias_h,
            po, so, stFbf, t, (t == Tv - 1) ? 1 : 0);
    }

    // OUT = state_T @ W_out^T + b_out
    gemm_bias_kernel<2048><<<dim3(NOUT / BN, Bv / BM), 256, 0, stream>>>(
        stFbf, Woutbf, b_out, (float*)d_out, Bv, NOUT);
}
